// CongestionAwareMP_71579924955602
// MI455X (gfx1250) — matrix operand order repaired and checked by reading, not compiled, not run on hardware
//
#include <hip/hip_runtime.h>

typedef __attribute__((ext_vector_type(16))) _Float16 v16h;
typedef __attribute__((ext_vector_type(8)))  _Float16 v8h;
typedef __attribute__((ext_vector_type(8)))  float    v8f;
typedef __attribute__((ext_vector_type(4)))  float    v4f;
typedef __attribute__((ext_vector_type(4)))  unsigned int v4u;

#define NNODES   100000
#define NEDGES   1600000
#define NODE_DIM 128
#define EDGE_DIM 32
#define HIDDEN   256
#define MSG_K    320        // 289 padded up to 10 K-chunks of 32
#define WAVES    8

// ---------------------------------------------------------------- helpers

__device__ __forceinline__ v8f wmma_f16f32(v16h a, v16h b, v8f c) {
  return __builtin_amdgcn_wmma_f32_16x16x32_f16(false, a, false, b, (short)0, c,
                                                false, false);
}

__device__ __forceinline__ v8f splat8(float x) {
  v8f r;
#pragma unroll
  for (int i = 0; i < 8; ++i) r[i] = x;
  return r;
}

// A fragment (16x32 f16): lane(0-15)=row r; K = kc*32 + {g*8..+7, 16+g*8..+7}
__device__ __forceinline__ v16h load_afrag(const _Float16* rowp, int kc, int g) {
  const _Float16* p = rowp + kc * 32 + g * 8;
  union { v16h v; v8h h[2]; } u;
  u.h[0] = *(const v8h*)(p);
  u.h[1] = *(const v8h*)(p + 16);
  return u.v;
}

// B fragment (32x16 f16): lane holds column n; K = kc*32 + {g*8..+7, 16+g*8..+7}
// Weights stored column-major f16: w[n*Kpad + k]
__device__ __forceinline__ v16h load_bfrag(const _Float16* w, int Kpad, int ncol,
                                           int kc, int g) {
  const _Float16* p = w + (size_t)ncol * Kpad + kc * 32 + g * 8;
  union { v16h v; v8h h[2]; } u;
  u.h[0] = *(const v8h*)(p);
  u.h[1] = *(const v8h*)(p + 16);
  return u.v;
}

__device__ __forceinline__ void cvt_store8(const float* src, _Float16* dst) {
  v4f f0 = *(const v4f*)(src);
  v4f f1 = *(const v4f*)(src + 4);
  v8h h;
#pragma unroll
  for (int j = 0; j < 4; ++j) { h[j] = (_Float16)f0[j]; h[4 + j] = (_Float16)f1[j]; }
  *(v8h*)(dst) = h;
}

// ---------------------------------------------------------------- prep kernels

__global__ void pack_kernel(const float* __restrict__ W, _Float16* __restrict__ out,
                            int Kin, int Ncols, int Kpad) {
  int idx = blockIdx.x * blockDim.x + threadIdx.x;
  int total = Ncols * Kpad;
  if (idx >= total) return;
  int k = idx % Kpad;
  int n = idx / Kpad;
  float v = (k < Kin) ? W[(long)k * Ncols + n] : 0.f;
  out[idx] = (_Float16)v;
}

__global__ void zero_kernel(float* __restrict__ p, int n4) {
  int i = blockIdx.x * blockDim.x + threadIdx.x;
  if (i < n4) {
    v4f z;
#pragma unroll
    for (int j = 0; j < 4; ++j) z[j] = 0.f;
    ((v4f*)p)[i] = z;
  }
}

// ---------------------------------------------------------------- edge kernel
// LDS: W1 packed (81920 halfs, 160 KB) + per-wave [A-tile 16x320 | hid 16x256]
//      = 81920 + 8*(5120+4096) halfs = 155648 halfs = 311296 B

__global__ void __launch_bounds__(256) edge_kernel(
    const float* __restrict__ x, const int* __restrict__ ei,
    const float* __restrict__ ea, const float* __restrict__ cong,
    const _Float16* __restrict__ w1p,   // packed W1 (256 cols x 320 K) f16, LDS-cached
    const _Float16* __restrict__ w2p,   // packed W2 (128 cols x 256 K) f16, from global
    const float* __restrict__ mb1, const float* __restrict__ mb2,
    float* __restrict__ agg) {
  extern __shared__ _Float16 smem[];
  _Float16* w1s = smem;                               // 81920 halfs
  const int tid = threadIdx.x;
  const int wv = tid >> 5, lane = tid & 31;
  const int r = lane & 15, g = lane >> 4;
  _Float16* tile = smem + 81920 + wv * (5120 + 4096); // 16x320 A staging
  _Float16* hid  = tile + 5120;                       // 16x256 hidden staging

  {
    const v4u* gs = (const v4u*)w1p;                  // 163840 B = 10240 x 16 B
    v4u* sd = (v4u*)smem;
    for (int i = tid; i < 10240; i += 256) sd[i] = gs[i];
  }
  __syncthreads();

  const int NT = NEDGES / 16;                         // 100000 tiles, exact
  for (int t = blockIdx.x * WAVES + wv; t < NT; t += gridDim.x * WAVES) {
    // ---- stage msg_in tile [16][320] f16: [x[src] | x[dst] | edge_attr | cong | 0]
    const long e = (long)t * 16 + r;
    const int s = ei[e];
    const int d = ei[NEDGES + e];
    const float* xr = x + (long)(g ? d : s) * NODE_DIM;
    _Float16* row = tile + r * MSG_K;
    _Float16* dst0 = row + g * NODE_DIM;
#pragma unroll
    for (int c = 0; c < NODE_DIM; c += 8) cvt_store8(xr + c, dst0 + c);
    if (!g) {
      const float* eap = ea + e * EDGE_DIM;
#pragma unroll
      for (int c = 0; c < EDGE_DIM; c += 8) cvt_store8(eap + c, row + 2 * NODE_DIM + c);
    } else {
      v8h z;
#pragma unroll
      for (int j = 0; j < 8; ++j) z[j] = (_Float16)0.f;
      *(v8h*)(row + 288) = z; *(v8h*)(row + 296) = z;
      *(v8h*)(row + 304) = z; *(v8h*)(row + 312) = z;
      row[288] = (_Float16)cong[s];
    }
    asm volatile("s_wait_dscnt 0" ::: "memory");

    // ---- MLP1: [16x320] @ [320x256]; 4 N-blocks at a time (32 acc VGPRs live)
#pragma unroll 1
    for (int nbc = 0; nbc < 16; nbc += 4) {
      v8f acc[4];
#pragma unroll
      for (int j = 0; j < 4; ++j) acc[j] = splat8(mb1[(nbc + j) * 16 + r]);
#pragma unroll
      for (int kc = 0; kc < 10; ++kc) {
        v16h a = load_afrag(tile + r * MSG_K, kc, g);
#pragma unroll
        for (int j = 0; j < 4; ++j)
          acc[j] = wmma_f16f32(a, load_bfrag(w1s, MSG_K, (nbc + j) * 16 + r, kc, g),
                               acc[j]);
      }
#pragma unroll
      for (int j = 0; j < 4; ++j) {
#pragma unroll
        for (int rr = 0; rr < 8; ++rr) {
          float v = acc[j][rr];
          v = v > 0.f ? v : 0.f;
          hid[(g * 8 + rr) * HIDDEN + (nbc + j) * 16 + r] = (_Float16)v;
        }
      }
    }
    asm volatile("s_wait_dscnt 0" ::: "memory");

    // destination nodes for this tile's rows (hoisted for both MLP2 chunks)
    int dn[8];
    const long eb2 = (long)t * 16 + g * 8;
#pragma unroll
    for (int rr = 0; rr < 8; ++rr) dn[rr] = ei[NEDGES + eb2 + rr];

    // ---- MLP2: [16x256] @ [256x128]; W2 B-frags streamed from global (L0/L2)
#pragma unroll 1
    for (int nbc = 0; nbc < 8; nbc += 4) {
      v8f acc[4];
#pragma unroll
      for (int j = 0; j < 4; ++j) acc[j] = splat8(mb2[(nbc + j) * 16 + r]);
#pragma unroll
      for (int kc = 0; kc < 8; ++kc) {
        v16h a = load_afrag(hid + r * HIDDEN, kc, g);
#pragma unroll
        for (int j = 0; j < 4; ++j)
          acc[j] = wmma_f16f32(a, load_bfrag(w2p, HIDDEN, (nbc + j) * 16 + r, kc, g),
                               acc[j]);
      }
      // scatter-add this chunk (fp32 atomics -> exact segment sum)
#pragma unroll
      for (int rr = 0; rr < 8; ++rr) {
        float* arow = agg + (long)dn[rr] * NODE_DIM + r;
#pragma unroll
        for (int j = 0; j < 4; ++j)
          unsafeAtomicAdd(arow + (nbc + j) * 16, acc[j][rr]);
      }
    }
  }
}

// ---------------------------------------------------------------- node kernel
// LDS: uW1 packed (65536 halfs, 128 KB) + per-wave [A-tile 16x256 | hid 16x256]
//      = 65536 + 8*(4096+4096) halfs = 131072 halfs = 262144 B

__global__ void __launch_bounds__(256) node_kernel(
    const float* __restrict__ x,
    const _Float16* __restrict__ u1p,   // packed uW1 (256 x 256) f16, LDS-cached
    const _Float16* __restrict__ u2p,   // packed uW2 (128 x 256) f16, from global
    const float* __restrict__ ub1, const float* __restrict__ ub2,
    float* __restrict__ io) {           // d_out: aggregated in, x_new out (in place)
  extern __shared__ _Float16 smem[];
  _Float16* u1s = smem;                               // 65536 halfs
  const int tid = threadIdx.x;
  const int wv = tid >> 5, lane = tid & 31;
  const int r = lane & 15, g = lane >> 4;
  _Float16* tile = smem + 65536 + wv * (4096 + 4096);
  _Float16* hid  = tile + 4096;

  {
    const v4u* gs = (const v4u*)u1p;                  // 131072 B = 8192 x 16 B
    v4u* sd = (v4u*)smem;
    for (int i = tid; i < 8192; i += 256) sd[i] = gs[i];
  }
  __syncthreads();

  const int NT = NNODES / 16;                         // 6250 tiles, exact
  for (int t = blockIdx.x * WAVES + wv; t < NT; t += gridDim.x * WAVES) {
    // ---- stage upd_in tile [16][256] f16: [x[n] | aggregated[n]]
    const long n = (long)t * 16 + r;
    const float* srcp = g ? (io + n * NODE_DIM) : (x + n * NODE_DIM);
    _Float16* dst0 = tile + r * HIDDEN + g * NODE_DIM;
#pragma unroll
    for (int c = 0; c < NODE_DIM; c += 8) cvt_store8(srcp + c, dst0 + c);
    asm volatile("s_wait_dscnt 0" ::: "memory");

    // ---- MLP1: [16x256] @ [256x256]; 4 N-blocks at a time
#pragma unroll 1
    for (int nbc = 0; nbc < 16; nbc += 4) {
      v8f acc[4];
#pragma unroll
      for (int j = 0; j < 4; ++j) acc[j] = splat8(ub1[(nbc + j) * 16 + r]);
#pragma unroll
      for (int kc = 0; kc < 8; ++kc) {
        v16h a = load_afrag(tile + r * HIDDEN, kc, g);
#pragma unroll
        for (int j = 0; j < 4; ++j)
          acc[j] = wmma_f16f32(a, load_bfrag(u1s, HIDDEN, (nbc + j) * 16 + r, kc, g),
                               acc[j]);
      }
#pragma unroll
      for (int j = 0; j < 4; ++j) {
#pragma unroll
        for (int rr = 0; rr < 8; ++rr) {
          float v = acc[j][rr];
          v = v > 0.f ? v : 0.f;
          hid[(g * 8 + rr) * HIDDEN + (nbc + j) * 16 + r] = (_Float16)v;
        }
      }
    }
    asm volatile("s_wait_dscnt 0" ::: "memory");

    // ---- MLP2: [16x256] @ [256x128]; uW2 B-frags from global
    const long n0 = (long)t * 16 + g * 8;
#pragma unroll 1
    for (int nbc = 0; nbc < 8; nbc += 4) {
      v8f acc[4];
#pragma unroll
      for (int j = 0; j < 4; ++j) acc[j] = splat8(ub2[(nbc + j) * 16 + r]);
#pragma unroll
      for (int kc = 0; kc < 8; ++kc) {
        v16h a = load_afrag(hid + r * HIDDEN, kc, g);
#pragma unroll
        for (int j = 0; j < 4; ++j)
          acc[j] = wmma_f16f32(a, load_bfrag(u2p, HIDDEN, (nbc + j) * 16 + r, kc, g),
                               acc[j]);
      }
      // write x_new over aggregated rows (this wave already consumed them)
#pragma unroll
      for (int rr = 0; rr < 8; ++rr) {
        float* orow = io + (n0 + rr) * NODE_DIM + r;
#pragma unroll
        for (int j = 0; j < 4; ++j) orow[(nbc + j) * 16] = acc[j][rr];
      }
    }
  }
}

// ---------------------------------------------------------------- launch

extern "C" void kernel_launch(void* const* d_in, const int* in_sizes, int n_in,
                              void* d_out, int out_size, void* d_ws, size_t ws_size,
                              hipStream_t stream) {
  (void)in_sizes; (void)n_in; (void)out_size; (void)ws_size;
  const float* x    = (const float*)d_in[0];
  const int*   ei   = (const int*)d_in[1];
  const float* ea   = (const float*)d_in[2];
  const float* cong = (const float*)d_in[3];
  const float* mW1  = (const float*)d_in[4];
  const float* mb1  = (const float*)d_in[5];
  const float* mW2  = (const float*)d_in[6];
  const float* mb2  = (const float*)d_in[7];
  const float* uW1  = (const float*)d_in[8];
  const float* ub1  = (const float*)d_in[9];
  const float* uW2  = (const float*)d_in[10];
  const float* ub2  = (const float*)d_in[11];
  float* out = (float*)d_out;

  char* ws = (char*)d_ws;                              // 425984 B of f16 weights
  _Float16* w1p = (_Float16*)(ws);                     // 256 x 320  (163840 B)
  _Float16* w2p = (_Float16*)(ws + 163840);            // 128 x 256  ( 65536 B)
  _Float16* u1p = (_Float16*)(ws + 229376);            // 256 x 256  (131072 B)
  _Float16* u2p = (_Float16*)(ws + 360448);            // 128 x 256  ( 65536 B)

  pack_kernel<<<(81920 + 255) / 256, 256, 0, stream>>>(mW1, w1p, 289, 256, 320);
  pack_kernel<<<(32768 + 255) / 256, 256, 0, stream>>>(mW2, w2p, 256, 128, 256);
  pack_kernel<<<(65536 + 255) / 256, 256, 0, stream>>>(uW1, u1p, 256, 256, 256);
  pack_kernel<<<(32768 + 255) / 256, 256, 0, stream>>>(uW2, u2p, 256, 128, 256);

  const int n4 = (NNODES * NODE_DIM) / 4;
  zero_kernel<<<(n4 + 255) / 256, 256, 0, stream>>>(out, n4);

  edge_kernel<<<600, 256, 311296, stream>>>(x, ei, ea, cong, w1p, w2p, mb1, mb2, out);
  node_kernel<<<200, 256, 262144, stream>>>(x, u1p, u2p, ub1, ub2, out);
}
